// JointAttentionBlock_55190329753678
// MI455X (gfx1250) — matrix operand order repaired and checked by reading, not compiled, not run on hardware
//
#include <hip/hip_runtime.h>
#include <math.h>

// ---------------------------------------------------------------------------
// JointAttentionBlock for MI455X (gfx1250):
//   bf16 WMMA + async global->LDS DMA (ASYNCcnt) + Tensor Data Mover (TENSORcnt)
// n=4, h=w=32 (s=1024), d_model=768, 12 heads x 64, kv_len=2048 (self+src).
// ---------------------------------------------------------------------------

#define N_B   4
#define SEQ   1024
#define DM    768
#define NH    12
#define DH    64
#define RTOT  (N_B*SEQ)   // 4096 tokens
#define KV    2048        // concatenated kv length

typedef __attribute__((ext_vector_type(16))) __bf16        v16bf;
typedef __attribute__((ext_vector_type(8)))  float         v8f;
typedef __attribute__((ext_vector_type(4)))  unsigned int  u32x4;
typedef __attribute__((ext_vector_type(4)))  float         f32x4;
typedef __attribute__((ext_vector_type(4)))  unsigned int  tdm_uv4;
typedef __attribute__((ext_vector_type(8)))  int           tdm_iv8;
typedef __attribute__((ext_vector_type(4)))  int           tdm_iv4;

union FragBF {
  v16bf v;
  unsigned short u[16];
  u32x4 q[2];
};
union Pack8 {
  u32x4 q;
  unsigned short u[8];
};

__device__ __forceinline__ unsigned short f2bf(float f) {
  unsigned int u = __float_as_uint(f);
  u += 0x7FFFu + ((u >> 16) & 1u);          // round-to-nearest-even
  return (unsigned short)(u >> 16);
}
__device__ __forceinline__ float bf2f(unsigned short h) {
  return __uint_as_float(((unsigned int)h) << 16);
}
__device__ __forceinline__ float wave_sum(float v) {
  #pragma unroll
  for (int o = 16; o > 0; o >>= 1) v += __shfl_xor(v, o, 32);
  return v;
}

// CDNA5 async DMA: global -> LDS, 16 bytes per lane, tracked with ASYNCcnt.
__device__ __forceinline__ void async_b128(void* lds, const void* gptr) {
  unsigned l = (unsigned)(unsigned long long)(uintptr_t)lds;   // LDS byte offset
  unsigned long long g = (unsigned long long)(uintptr_t)gptr;
  asm volatile("global_load_async_to_lds_b128 %0, %1, off"
               :: "v"(l), "v"(g) : "memory");
}
__device__ __forceinline__ void wait_async0() {
  asm volatile("s_wait_asynccnt 0" ::: "memory");
}

// CDNA5 Tensor Data Mover: 2D tile (bf16 elements) global -> LDS, TENSORcnt.
// Descriptor bitfields per CDNA5 ISA ch.8 (D# group0 / group1).
__device__ __forceinline__ void tdm_load_2d(unsigned ldsAddr, unsigned long long gaddr,
                                            int tileW, int tileH, int rowStride,
                                            int padInterval, int padAmount, int padEnable) {
  tdm_uv4 g0;
  g0[0] = 1u;                                            // count=1 (valid user D#)
  g0[1] = ldsAddr;                                       // lds_addr
  g0[2] = (unsigned)gaddr;                               // global_addr[31:0]
  g0[3] = (unsigned)((gaddr >> 32) & 0x01FFFFFFull)      // global_addr[56:32]
        | 0x80000000u;                                   // type=2 ("image")
  tdm_iv8 g1;
  unsigned w0 = (1u << 16);                              // data_size=1 -> 2 bytes
  if (padEnable)
    w0 |= (1u << 20) | ((unsigned)padInterval << 22) | ((unsigned)padAmount << 25);
  g1[0] = (int)w0;                                       // wg_mask=0 (not in cluster)
  g1[1] = (int)((unsigned)tileW << 16);                  // tensor_dim0[15:0]
  g1[2] = (int)(((unsigned)tileW >> 16) | ((unsigned)tileH << 16)); // dim0 hi | tensor_dim1 lo
  g1[3] = (int)(((unsigned)tileH >> 16) | ((unsigned)tileW << 16)); // dim1 hi | tile_dim0
  g1[4] = (int)(unsigned)tileH;                          // tile_dim1 (tile_dim2=0)
  g1[5] = rowStride;                                     // tensor_dim0_stride[31:0]
  g1[6] = 0;                                             // stride hi | dim1_stride lo
  g1[7] = 0;
  tdm_iv4 z4 = {0, 0, 0, 0};
#if __clang_major__ >= 23
  tdm_iv8 z8 = {0, 0, 0, 0, 0, 0, 0, 0};                 // amdgpu-toolchain: 6-arg form
  __builtin_amdgcn_tensor_load_to_lds(g0, g1, z4, z4, z8, 0);
#else
  __builtin_amdgcn_tensor_load_to_lds(g0, g1, z4, z4, 0); // ROCm 7.2: 5-arg form
#endif
}
__device__ __forceinline__ void wait_tensor0() {
  __builtin_amdgcn_s_wait_tensorcnt(0);
}

// ---------------------------------------------------------------- fp32->bf16
__global__ void k_f32_to_bf16(const float* __restrict__ src,
                              unsigned short* __restrict__ dst, int count) {
  int i = (blockIdx.x * blockDim.x + threadIdx.x) * 8;
  if (i >= count) return;
  f32x4 a = *(const f32x4*)(src + i);
  f32x4 b = *(const f32x4*)(src + i + 4);
  Pack8 p;
  #pragma unroll
  for (int j = 0; j < 4; j++) { p.u[j] = f2bf(a[j]); p.u[4 + j] = f2bf(b[j]); }
  *(u32x4*)(dst + i) = p.q;
}

// ------------------------------------------------- adaLN: silu(cond)@ada_w^T
__global__ void k_ada(const float* __restrict__ cond,
                      const float* __restrict__ ada_w,
                      const float* __restrict__ ada_b,
                      float* __restrict__ ada) {
  int gw   = blockIdx.x * (blockDim.x >> 5) + (threadIdx.x >> 5);
  int lane = threadIdx.x & 31;
  if (gw >= N_B * 3 * DM) return;
  int nb = gw / (3 * DM), o = gw % (3 * DM);
  float p = 0.f;
  for (int k = lane; k < DM; k += 32) {
    float c = cond[nb * DM + k];
    float s = c / (1.f + __expf(-c));        // silu
    p += s * ada_w[(size_t)o * DM + k];
  }
  p = wave_sum(p);
  if (lane == 0) ada[nb * 3 * DM + o] = p + ada_b[o];
}

// ------------------------------------- RMS-norm + (1+scale)*x + shift -> bf16
__global__ void k_mod(const float* __restrict__ x, const float* __restrict__ src,
                      const float* __restrict__ ada,
                      const float* __restrict__ qns, const float* __restrict__ sns,
                      unsigned short* __restrict__ xm, unsigned short* __restrict__ sm) {
  int tok  = blockIdx.x * (blockDim.x >> 5) + (threadIdx.x >> 5);
  int lane = threadIdx.x & 31;
  if (tok >= RTOT) return;
  int nb = tok / SEQ;
  const float* xp = x   + (size_t)tok * DM;
  const float* sp = src + (size_t)tok * DM;
  float s2x = 0.f, s2s = 0.f;
  for (int d = lane; d < DM; d += 32) {
    float a = xp[d]; s2x += a * a;
    float b = sp[d]; s2s += b * b;
  }
  s2x = wave_sum(s2x); s2s = wave_sum(s2s);
  float ix = rsqrtf(s2x / (float)DM + 1e-6f);
  float is = rsqrtf(s2s / (float)DM + 1e-6f);
  const float* ap = ada + (size_t)nb * 3 * DM;
  for (int d = lane; d < DM; d += 32) {
    float sc = 1.f + ap[DM + d], sh = ap[d];
    xm[(size_t)tok * DM + d] = f2bf(xp[d] * qns[d] * ix * sc + sh);
    sm[(size_t)tok * DM + d] = f2bf(sp[d] * sns[d] * is * sc + sh);
  }
}

// --------------------------------------------------------------- WMMA GEMM
// C[r,c] = sum_k A[r,k] * W[c,k]   (A: 4096x768 bf16, W: 768x768 bf16)
// block = 256 threads = 8 waves; wave tile 16x32 (one A frag, two B frags)
// block tile 32 rows x 128 cols; K staged 64 at a time via async DMA.
__global__ __launch_bounds__(256)
void k_gemm(const unsigned short* __restrict__ A, const unsigned short* __restrict__ W,
            unsigned short* __restrict__ outBf, float* __restrict__ outF,
            const float* __restrict__ skip, const float* __restrict__ gate,
            int outBatchStride, int outRowOffset) {
  __shared__ __align__(16) unsigned short As[32][72];
  __shared__ __align__(16) unsigned short Ws[128][72];
  int tid = threadIdx.x, lane = tid & 31, wv = tid >> 5;
  int wr = wv >> 2, wc = wv & 3;                  // wave row tile / col pair
  int rowBase = blockIdx.x * 32, colBase = blockIdx.y * 128;
  int m = lane & 15, hi = lane >> 4;
  v8f acc0 = {}, acc1 = {};
  for (int k0 = 0; k0 < DM; k0 += 64) {
    __syncthreads();
    {
      // A tile: 32 rows x 64 cols = 256 x 16B chunks (1 per thread)
      int r = tid >> 3, c = (tid & 7) * 8;
      async_b128(&As[r][c], &A[(size_t)(rowBase + r) * DM + k0 + c]);
      // W tile: 128 rows x 64 cols = 1024 chunks (4 per thread)
      #pragma unroll
      for (int j = 0; j < 4; j++) {
        int i  = tid * 4 + j;
        int r0 = i >> 3, c0 = (i & 7) * 8;
        async_b128(&Ws[r0][c0], &W[(size_t)(colBase + r0) * DM + k0 + c0]);
      }
    }
    if (k0 + 64 < DM) {
      __builtin_prefetch(&A[(size_t)rowBase * DM + k0 + 64]);
      __builtin_prefetch(&W[(size_t)colBase * DM + k0 + 64]);
    }
    wait_async0();
    __syncthreads();
    #pragma unroll
    for (int s = 0; s < 2; s++) {
      FragBF a, b0, b1;
      const unsigned short* ap = &As[wr * 16 + m][s * 32];
      a.q[0] = *(const u32x4*)(ap + hi * 16);       // K = hi*16 .. +7
      a.q[1] = *(const u32x4*)(ap + hi * 16 + 8);   // K = hi*16+8 .. +15
      const unsigned short* bp0 = &Ws[wc * 32 + m][s * 32 + hi * 16];
      b0.q[0] = *(const u32x4*)(bp0); b0.q[1] = *(const u32x4*)(bp0 + 8);
      const unsigned short* bp1 = &Ws[wc * 32 + 16 + m][s * 32 + hi * 16];
      b1.q[0] = *(const u32x4*)(bp1); b1.q[1] = *(const u32x4*)(bp1 + 8);
      acc0 = __builtin_amdgcn_wmma_f32_16x16x32_bf16(false, a.v, false, b0.v,
                                                     (short)0, acc0, false, false);
      acc1 = __builtin_amdgcn_wmma_f32_16x16x32_bf16(false, a.v, false, b1.v,
                                                     (short)0, acc1, false, false);
    }
  }
  int col0 = colBase + wc * 32 + m, col1 = col0 + 16;
  #pragma unroll
  for (int v = 0; v < 8; v++) {
    int r  = rowBase + wr * 16 + v + 8 * hi;        // D row = vgpr + 8*(lane>=16)
    int nb = r / SEQ;
    size_t orow = (size_t)nb * outBatchStride + (r % SEQ) + outRowOffset;
    if (outBf) {
      outBf[orow * DM + col0] = f2bf(acc0[v]);
      outBf[orow * DM + col1] = f2bf(acc1[v]);
    } else {
      float g0 = gate[(size_t)nb * 3 * DM + 2 * DM + col0];
      float g1 = gate[(size_t)nb * 3 * DM + 2 * DM + col1];
      outF[orow * DM + col0] = skip[(size_t)r * DM + col0] + g0 * acc0[v];
      outF[orow * DM + col1] = skip[(size_t)r * DM + col1] + g1 * acc1[v];
    }
  }
}

// --------------------------------------- cosine scale + axial RoPE (bf16 io)
__global__ void k_rope(const unsigned short* __restrict__ in,
                       unsigned short* __restrict__ out,
                       const float* __restrict__ pos,
                       const float* __restrict__ attn_scale,
                       int outBatchStride, int outRowOffset) {
  int id = blockIdx.x * blockDim.x + threadIdx.x;
  if (id >= RTOT * NH) return;
  int head = id % NH, tok = id / NH;
  int nb = tok / SEQ, s = tok % SEQ;
  const u32x4* sp4 = (const u32x4*)(in + (size_t)tok * DM + head * DH);
  float t[64], o[64];
  float ssum = 0.f;
  #pragma unroll
  for (int i = 0; i < 8; i++) {
    Pack8 p; p.q = sp4[i];
    #pragma unroll
    for (int j = 0; j < 8; j++) { float v = bf2f(p.u[j]); t[i * 8 + j] = v; ssum += v * v; }
  }
  float scl = sqrtf(attn_scale[head]) * rsqrtf(ssum + 1e-6f);
  #pragma unroll
  for (int i = 0; i < 64; i++) t[i] *= scl;
  float p0 = pos[(size_t)tok * 2 + 0], p1 = pos[(size_t)tok * 2 + 1];
  const float LOGPI = 1.1447298858494002f;
  const float STEP  = 2.302585092994046f / 96.f;   // log(10)/96
  #pragma unroll
  for (int j = 0; j < 8; j++) {
    float fr = __expf(LOGPI + (float)(j * NH + head) * STEP);
    float th = p0 * fr, tw = p1 * fr;
    float ch = __cosf(th), shh = __sinf(th);
    float cw = __cosf(tw), sw  = __sinf(tw);
    o[j]      = t[j]      * ch - t[16 + j] * shh;    // theta_h dims
    o[16 + j] = t[16 + j] * ch + t[j]      * shh;
    o[8 + j]  = t[8 + j]  * cw - t[24 + j] * sw;     // theta_w dims
    o[24 + j] = t[24 + j] * cw + t[8 + j]  * sw;
  }
  #pragma unroll
  for (int i = 32; i < 64; i++) o[i] = t[i];
  size_t orow = (size_t)nb * outBatchStride + s + outRowOffset;
  u32x4* dp4 = (u32x4*)(out + orow * DM + head * DH);
  #pragma unroll
  for (int i = 0; i < 8; i++) {
    Pack8 p;
    #pragma unroll
    for (int j = 0; j < 8; j++) p.u[j] = f2bf(o[i * 8 + j]);
    dp4[i] = p.q;
  }
}

// ------------------------------------------- flash attention (WMMA, bf16 in)
// grid = n*heads blocks; 8 waves; wave handles 8 q-tiles of 16 rows.
// K tile staged by the Tensor Data Mover (wave 0), V tile transposed manually.
__global__ __launch_bounds__(256)
void k_attn(const unsigned short* __restrict__ q,
            const unsigned short* __restrict__ kall,
            const unsigned short* __restrict__ vall,
            unsigned short* __restrict__ obuf) {
  __shared__ __align__(16) unsigned short Ks[32][72];   // [key][dim] (TDM-padded)
  __shared__ __align__(16) unsigned short Vt[64][40];   // [dim][key] transposed
  __shared__ __align__(16) float          scS[8][16][33];
  __shared__ __align__(16) unsigned short pS[8][16][32];
  __shared__ __align__(16) float          rowS[8][16];

  int tid = threadIdx.x, lane = tid & 31, wv = tid >> 5;
  int nb = blockIdx.x / NH, head = blockIdx.x % NH;
  int m = lane & 15, hi = lane >> 4;

  const unsigned short* qb = q    + (size_t)nb * SEQ * DM + head * DH;
  const unsigned short* kb = kall + (size_t)nb * KV  * DM + head * DH;
  const unsigned short* vb = vall + (size_t)nb * KV  * DM + head * DH;

  for (int qt = 0; qt < 8; qt++) {
    int qrow0 = wv * 128 + qt * 16;
    FragBF qa0, qa1;
    {
      const unsigned short* qp = qb + (size_t)(qrow0 + m) * DM;
      qa0.q[0] = *(const u32x4*)(qp + hi * 16);
      qa0.q[1] = *(const u32x4*)(qp + 8 + hi * 16);
      qa1.q[0] = *(const u32x4*)(qp + 32 + hi * 16);
      qa1.q[1] = *(const u32x4*)(qp + 40 + hi * 16);
    }
    float mstate = -3.0e38f, lstate = 0.f;
    v8f acc0 = {}, acc1 = {}, acc2 = {}, acc3 = {};
    for (int ch = 0; ch < KV / 32; ch++) {
      int key0 = ch * 32;
      __syncthreads();
      if (wv == 0) {
        // K chunk via Tensor Data Mover: 32 rows x 64 bf16, row stride 768,
        // LDS pad 4 dwords (16B) after every 32 dwords (128B) -> Ks[.][72].
        tdm_load_2d((unsigned)(uintptr_t)&Ks[0][0],
                    (unsigned long long)(uintptr_t)&kb[(size_t)key0 * DM],
                    64, 32, DM, /*padInterval(32dw)=*/4, /*padAmount(4dw)=*/3, 1);
      }
      {
        // V chunk transposed: b128 load + 8 b16 LDS stores per thread
        int key = tid & 31, dg = tid >> 5;          // dg = 0..7
        Pack8 p;
        p.q = *(const u32x4*)&vb[(size_t)(key0 + key) * DM + dg * 8];
        #pragma unroll
        for (int j = 0; j < 8; j++) Vt[dg * 8 + j][key] = p.u[j];
      }
      if (wv == 0) wait_tensor0();
      __syncthreads();
      // ---- logits: 16 q-rows x 32 keys, reduce over dhead=64 (2 wmma steps)
      v8f c0 = {}, c1 = {};
      {
        FragBF b;
        const unsigned short* bp = &Ks[m][hi * 16];
        b.q[0] = *(const u32x4*)bp; b.q[1] = *(const u32x4*)(bp + 8);
        c0 = __builtin_amdgcn_wmma_f32_16x16x32_bf16(false, qa0.v, false, b.v, (short)0, c0, false, false);
        bp = &Ks[m][32 + hi * 16];
        b.q[0] = *(const u32x4*)bp; b.q[1] = *(const u32x4*)(bp + 8);
        c0 = __builtin_amdgcn_wmma_f32_16x16x32_bf16(false, qa1.v, false, b.v, (short)0, c0, false, false);
        bp = &Ks[16 + m][hi * 16];
        b.q[0] = *(const u32x4*)bp; b.q[1] = *(const u32x4*)(bp + 8);
        c1 = __builtin_amdgcn_wmma_f32_16x16x32_bf16(false, qa0.v, false, b.v, (short)0, c1, false, false);
        bp = &Ks[16 + m][32 + hi * 16];
        b.q[0] = *(const u32x4*)bp; b.q[1] = *(const u32x4*)(bp + 8);
        c1 = __builtin_amdgcn_wmma_f32_16x16x32_bf16(false, qa1.v, false, b.v, (short)0, c1, false, false);
      }
      #pragma unroll
      for (int v = 0; v < 8; v++) {                       // C layout -> LDS
        scS[wv][v + 8 * hi][m]      = c0[v];
        scS[wv][v + 8 * hi][16 + m] = c1[v];
      }
      asm volatile("s_wait_dscnt 0" ::: "memory");
      if (lane < 16) {                                    // online softmax rows
        int row = lane;
        float mx = mstate;
        #pragma unroll
        for (int c = 0; c < 32; c++) mx = fmaxf(mx, scS[wv][row][c]);
        float corr = __expf(mstate - mx);
        float sum = lstate * corr;
        #pragma unroll
        for (int c = 0; c < 32; c++) {
          float p = __expf(scS[wv][row][c] - mx);
          pS[wv][row][c] = f2bf(p);
          sum += p;
        }
        mstate = mx; lstate = sum;
        rowS[wv][row] = corr;
      }
      asm volatile("s_wait_dscnt 0" ::: "memory");
      #pragma unroll
      for (int v = 0; v < 8; v++) {                       // rescale running o
        float rc = rowS[wv][v + 8 * hi];
        acc0[v] *= rc; acc1[v] *= rc; acc2[v] *= rc; acc3[v] *= rc;
      }
      // ---- o += p(16x32) @ V(32x64)
      FragBF pa, b;
      const unsigned short* pp = &pS[wv][m][0];
      pa.q[0] = *(const u32x4*)(pp + hi * 16);
      pa.q[1] = *(const u32x4*)(pp + 8 + hi * 16);
      const unsigned short* vp = &Vt[0 * 16 + m][hi * 16];
      b.q[0] = *(const u32x4*)vp; b.q[1] = *(const u32x4*)(vp + 8);
      acc0 = __builtin_amdgcn_wmma_f32_16x16x32_bf16(false, pa.v, false, b.v, (short)0, acc0, false, false);
      vp = &Vt[1 * 16 + m][hi * 16];
      b.q[0] = *(const u32x4*)vp; b.q[1] = *(const u32x4*)(vp + 8);
      acc1 = __builtin_amdgcn_wmma_f32_16x16x32_bf16(false, pa.v, false, b.v, (short)0, acc1, false, false);
      vp = &Vt[2 * 16 + m][hi * 16];
      b.q[0] = *(const u32x4*)vp; b.q[1] = *(const u32x4*)(vp + 8);
      acc2 = __builtin_amdgcn_wmma_f32_16x16x32_bf16(false, pa.v, false, b.v, (short)0, acc2, false, false);
      vp = &Vt[3 * 16 + m][hi * 16];
      b.q[0] = *(const u32x4*)vp; b.q[1] = *(const u32x4*)(vp + 8);
      acc3 = __builtin_amdgcn_wmma_f32_16x16x32_bf16(false, pa.v, false, b.v, (short)0, acc3, false, false);
    }
    if (lane < 16) rowS[wv][lane] = 1.0f / lstate;
    asm volatile("s_wait_dscnt 0" ::: "memory");
    unsigned short* op = obuf + (size_t)(nb * SEQ + qrow0) * DM + head * DH;
    #pragma unroll
    for (int v = 0; v < 8; v++) {
      int r = v + 8 * hi;
      float inv = rowS[wv][r];
      op[(size_t)r * DM +  0 + m] = f2bf(acc0[v] * inv);
      op[(size_t)r * DM + 16 + m] = f2bf(acc1[v] * inv);
      op[(size_t)r * DM + 32 + m] = f2bf(acc2[v] * inv);
      op[(size_t)r * DM + 48 + m] = f2bf(acc3[v] * inv);
    }
  }
}

// ---------------------------------------------------------------------------
extern "C" void kernel_launch(void* const* d_in, const int* in_sizes, int n_in,
                              void* d_out, int out_size, void* d_ws, size_t ws_size,
                              hipStream_t stream) {
  const float* x    = (const float*)d_in[0];
  const float* src  = (const float*)d_in[1];
  const float* pos  = (const float*)d_in[2];
  const float* cond = (const float*)d_in[3];
  const float* q_w  = (const float*)d_in[4];
  const float* k_w  = (const float*)d_in[5];
  const float* v_w  = (const float*)d_in[6];
  const float* o_w  = (const float*)d_in[7];
  const float* ada_w = (const float*)d_in[8];
  const float* ada_b = (const float*)d_in[9];
  const float* qns  = (const float*)d_in[10];
  const float* sns  = (const float*)d_in[11];
  const float* asc  = (const float*)d_in[12];
  float* out = (float*)d_out;

  char* ws = (char*)d_ws;
  size_t off = 0;
  auto alloc = [&](size_t bytes) -> void* {
    void* p = ws + off;
    off = (off + bytes + 255) & ~(size_t)255;
    return p;
  };
  float*          adaF  = (float*)         alloc((size_t)N_B * 3 * DM * 4);
  unsigned short* xm    = (unsigned short*)alloc((size_t)RTOT * DM * 2);
  unsigned short* sm    = (unsigned short*)alloc((size_t)RTOT * DM * 2);
  unsigned short* wqB   = (unsigned short*)alloc((size_t)DM * DM * 2);
  unsigned short* wkB   = (unsigned short*)alloc((size_t)DM * DM * 2);
  unsigned short* wvB   = (unsigned short*)alloc((size_t)DM * DM * 2);
  unsigned short* woB   = (unsigned short*)alloc((size_t)DM * DM * 2);
  unsigned short* qRaw  = (unsigned short*)alloc((size_t)RTOT * DM * 2);
  unsigned short* qRot  = (unsigned short*)alloc((size_t)RTOT * DM * 2);
  unsigned short* ksRaw = (unsigned short*)alloc((size_t)RTOT * DM * 2);
  unsigned short* kxRaw = (unsigned short*)alloc((size_t)RTOT * DM * 2);
  unsigned short* kAll  = (unsigned short*)alloc((size_t)N_B * KV * DM * 2);
  unsigned short* vAll  = (unsigned short*)alloc((size_t)N_B * KV * DM * 2);
  unsigned short* oBuf  = (unsigned short*)alloc((size_t)RTOT * DM * 2);

  const int WCNT = DM * DM;                         // 589824 (divisible by 8)
  const int WB8  = (WCNT / 8 + 255) / 256;
  k_f32_to_bf16<<<WB8, 256, 0, stream>>>(q_w, wqB, WCNT);
  k_f32_to_bf16<<<WB8, 256, 0, stream>>>(k_w, wkB, WCNT);
  k_f32_to_bf16<<<WB8, 256, 0, stream>>>(v_w, wvB, WCNT);
  k_f32_to_bf16<<<WB8, 256, 0, stream>>>(o_w, woB, WCNT);

  k_ada<<<(N_B * 3 * DM) / 8, 256, 0, stream>>>(cond, ada_w, ada_b, adaF);
  k_mod<<<RTOT / 8, 256, 0, stream>>>(x, src, adaF, qns, sns, xm, sm);

  dim3 gg(RTOT / 32, DM / 128);                     // 128 x 6 blocks
  // projections (bf16 out)
  k_gemm<<<gg, 256, 0, stream>>>(xm, wqB, qRaw,  nullptr, nullptr, nullptr, SEQ, 0);
  k_gemm<<<gg, 256, 0, stream>>>(xm, wkB, ksRaw, nullptr, nullptr, nullptr, SEQ, 0);
  k_gemm<<<gg, 256, 0, stream>>>(sm, wkB, kxRaw, nullptr, nullptr, nullptr, SEQ, 0);
  k_gemm<<<gg, 256, 0, stream>>>(xm, wvB, vAll,  nullptr, nullptr, nullptr, KV, 0);    // v_self
  k_gemm<<<gg, 256, 0, stream>>>(sm, wvB, vAll,  nullptr, nullptr, nullptr, KV, SEQ);  // v_src

  const int RB = (RTOT * NH) / 256;                 // 192 blocks
  k_rope<<<RB, 256, 0, stream>>>(qRaw,  qRot, pos, asc, SEQ, 0);
  k_rope<<<RB, 256, 0, stream>>>(ksRaw, kAll, pos, asc, KV, 0);    // k_self
  k_rope<<<RB, 256, 0, stream>>>(kxRaw, kAll, pos, asc, KV, SEQ);  // k_src

  k_attn<<<N_B * NH, 256, 0, stream>>>(qRot, kAll, vAll, oBuf);

  // output projection + gated residual (fp32 out)
  k_gemm<<<gg, 256, 0, stream>>>(oBuf, woB, nullptr, out, x, adaF, SEQ, 0);
}
